// SSIMLoss_62680752718560
// MI455X (gfx1250) — matrix operand order repaired and checked by reading, not compiled, not run on hardware
//
#include <hip/hip_runtime.h>
#include <hip/hip_bf16.h>

typedef __attribute__((ext_vector_type(16))) _Float16 v16h;
typedef __attribute__((ext_vector_type(8)))  float    v8f;

#define IMG_H 512
#define IMG_W 512
#define TILE  32          // output tile per workgroup
#define HALO_DIM 48       // 32 + 16 halo/pad (rows: -8..+39, cols: -5..+42)
#define SX_STRIDE 49      // bank-conflict-free (gcd(49,64)=1)
#define T1_STRIDE 57      // bank-conflict-free (gcd(57,64)=1), j reads up to 50
#define T1_PER_Q  (32 * T1_STRIDE)

__global__ __launch_bounds__(256)
void ssim_tile_kernel(const float* __restrict__ X, const float* __restrict__ Y,
                      float* __restrict__ partials) {
    const int tid  = threadIdx.x;
    const int lane = tid & 31;
    const int wid  = tid >> 5;
    const int img  = blockIdx.z;
    const int c0   = blockIdx.x * TILE;
    const int r0   = blockIdx.y * TILE;

    __shared__ float sX[HALO_DIM * SX_STRIDE];
    __shared__ float sY[HALO_DIM * SX_STRIDE];
    __shared__ float sT1[5 * T1_PER_Q];
    __shared__ float sPart[256];

    const float* xp = X + img * (IMG_H * IMG_W);
    const float* yp = Y + img * (IMG_H * IMG_W);

    // ---- Gaussian taps (sigma = 1.5, 11 taps, normalized) ----
    float gw[11];
    float gs = 0.0f;
#pragma unroll
    for (int t = 0; t < 11; ++t) {
        float d = (float)(t - 5);
        float w = __expf(-d * d / 4.5f);
        gw[t] = w; gs += w;
    }
    const float ginv = 1.0f / gs;
#pragma unroll
    for (int t = 0; t < 11; ++t) gw[t] *= ginv;

    // ---- constant A fragment: Toeplitz band A[m][k] = g[k-m], k in [m, m+10] ----
    // f16 A k order used here (contiguous, not AMD's table order): lanes 0-15 row m=lane, elems 0..15 -> K0..15;
    // lanes 16-31 row m=lane&15, elems 0..15 -> K16..31 (matches B fills).
    const int nrow = lane & 15;
    const int kh   = (lane >> 4) & 1;
    union { v16h v; _Float16 h[16]; } A;
#pragma unroll
    for (int e = 0; e < 16; ++e) {
        int k = e + kh * 16;
        int t = k - nrow;
        A.h[e] = (t >= 0 && t <= 10) ? (_Float16)gw[t] : (_Float16)0.0f;
    }

    // ---- stage haloed x/y tiles into LDS (zero padding == reference conv pad) ----
    for (int i = tid; i < HALO_DIM * HALO_DIM; i += 256) {
        int r = i / HALO_DIM, c = i % HALO_DIM;
        int gr = r0 - 8 + r;
        int gc = c0 - 5 + c;
        bool ok = (gr >= 0) & (gr < IMG_H) & (gc >= 0) & (gc < IMG_W);
        int o = r * SX_STRIDE + c;
        sX[o] = ok ? xp[gr * IMG_W + gc] : 0.0f;
        sY[o] = ok ? yp[gr * IMG_W + gc] : 0.0f;
    }
    // zero intermediate (incl. pad cols read under zero A-weights -> NaN-safe)
    for (int i = tid; i < 5 * T1_PER_Q; i += 256) sT1[i] = 0.0f;
    __syncthreads();

    // ================= pass 1: horizontal conv, output transposed ================
    // D[m][n] = Hq[row rt*16+n][col ct*16+m]; B[k][n] = q[row][localcol ct*16+k]
    if (wid < 6) {
        const int ct = wid & 1, rt = wid >> 1;             // 2 col-tiles x 3 row-tiles
        const int lrow = rt * 16 + nrow;                   // LDS row (0..47)
        const int cb   = ct * 16 + kh * 16;                // k-half base column
        float xv[16], yv[16];
#pragma unroll
        for (int i = 0; i < 16; ++i) {
            xv[i] = sX[lrow * SX_STRIDE + cb + i];
            yv[i] = sY[lrow * SX_STRIDE + cb + i];
        }
        auto mac = [&](v8f acc, auto f) -> v8f {
            union { v16h v; _Float16 h[16]; } bh, bl;
#pragma unroll
            for (int i = 0; i < 16; ++i) {
                float fv = f(i);
                _Float16 hi = (_Float16)fv;
                bh.h[i] = hi;
                bl.h[i] = (_Float16)(fv - (float)hi);      // hi/lo split: ~22-bit inputs
            }
            acc = __builtin_amdgcn_wmma_f32_16x16x32_f16(false, A.v, false, bh.v,
                                                         (short)0, acc, false, false);
            acc = __builtin_amdgcn_wmma_f32_16x16x32_f16(false, A.v, false, bl.v,
                                                         (short)0, acc, false, false);
            return acc;
        };
        v8f aX = {}, aY = {}, aXX = {}, aYY = {}, aXY = {};
        aX  = mac(aX,  [&](int i) { return xv[i]; });
        aY  = mac(aY,  [&](int i) { return yv[i]; });
        aXX = mac(aXX, [&](int i) { return xv[i] * xv[i]; });
        aYY = mac(aYY, [&](int i) { return yv[i] * yv[i]; });
        aXY = mac(aXY, [&](int i) { return xv[i] * yv[i]; });
        // store transposed intermediate: T1[q][c = ct*16+m][j = rt*16+n]
        auto st = [&](v8f a, int q) {
#pragma unroll
            for (int v = 0; v < 8; ++v) {
                int m = v + kh * 8;                        // C/D layout: M = v + 8*(lane>=16)
                sT1[q * T1_PER_Q + (ct * 16 + m) * T1_STRIDE + rt * 16 + nrow] = a[v];
            }
        };
        st(aX, 0); st(aY, 1); st(aXX, 2); st(aYY, 3); st(aXY, 4);
    }
    __syncthreads();

    // ====== pass 2: vertical conv == horizontal conv on transposed T1 + SSIM ======
    float part = 0.0f;
    if (wid < 4) {
        const int ct2 = wid & 1, rt2 = wid >> 1;           // 2x2 tiles of 16x16 outputs
        const int ccol = ct2 * 16 + nrow;                  // T1 "row" = image column
        const int jb   = rt2 * 16 + 3 + kh * 16;           // j = rt2*16 + k + 3
        auto mac2 = [&](int q) -> v8f {
            union { v16h v; _Float16 h[16]; } bh, bl;
#pragma unroll
            for (int i = 0; i < 16; ++i) {
                float fv = sT1[q * T1_PER_Q + ccol * T1_STRIDE + jb + i];
                _Float16 hi = (_Float16)fv;
                bh.h[i] = hi;
                bl.h[i] = (_Float16)(fv - (float)hi);
            }
            v8f acc = {};
            acc = __builtin_amdgcn_wmma_f32_16x16x32_f16(false, A.v, false, bh.v,
                                                         (short)0, acc, false, false);
            acc = __builtin_amdgcn_wmma_f32_16x16x32_f16(false, A.v, false, bl.v,
                                                         (short)0, acc, false, false);
            return acc;
        };
        v8f mX  = mac2(0);
        v8f mY  = mac2(1);
        v8f eXX = mac2(2);
        v8f eYY = mac2(3);
        v8f eXY = mac2(4);
        const float c1 = 1e-4f, c2 = 9e-4f;                // (0.01*R)^2, (0.03*R)^2
#pragma unroll
        for (int v = 0; v < 8; ++v) {
            float mux = mX[v],  muy = mY[v];
            float mux2 = mux * mux, muy2 = muy * muy, muxy = mux * muy;
            float sx2 = eXX[v] - mux2;
            float sy2 = eYY[v] - muy2;
            float sxy = eXY[v] - muxy;
            float num = (2.0f * muxy + c1) * (2.0f * sxy + c2);
            float den = (mux2 + muy2 + c1) * (sx2 + sy2 + c2) + 1e-8f;
            part += num / den;
        }
    }
    // deterministic per-block reduction (fixed order, no float atomics)
    sPart[tid] = part;
    __syncthreads();
    if (tid == 0) {
        float s = 0.0f;
        for (int i = 0; i < 256; ++i) s += sPart[i];
        partials[(blockIdx.z * gridDim.y + blockIdx.y) * gridDim.x + blockIdx.x] = s;
    }
}

__global__ __launch_bounds__(256)
void ssim_reduce_kernel(const float* __restrict__ partials, float* __restrict__ out,
                        int nparts, float inv_n) {
    __shared__ float sp[256];
    float s = 0.0f;
    for (int i = threadIdx.x; i < nparts; i += 256) s += partials[i];
    sp[threadIdx.x] = s;
    __syncthreads();
    if (threadIdx.x == 0) {
        float t = 0.0f;
        for (int i = 0; i < 256; ++i) t += sp[i];
        out[0] = 1.0f - t * inv_n;
    }
}

extern "C" void kernel_launch(void* const* d_in, const int* in_sizes, int n_in,
                              void* d_out, int out_size, void* d_ws, size_t ws_size,
                              hipStream_t stream) {
    const float* x = (const float*)d_in[0];
    const float* y = (const float*)d_in[1];
    float* out   = (float*)d_out;
    float* parts = (float*)d_ws;

    const int n_elems = in_sizes[0];                 // B*C*H*W = 12,582,912
    const int n_img   = n_elems / (IMG_H * IMG_W);   // 48
    dim3 grid(IMG_W / TILE, IMG_H / TILE, n_img);    // 16 x 16 x 48
    const int nparts = grid.x * grid.y * grid.z;

    ssim_tile_kernel<<<grid, dim3(256), 0, stream>>>(x, y, parts);
    ssim_reduce_kernel<<<1, 256, 0, stream>>>(parts, out, nparts, 1.0f / (float)n_elems);
}
